// DenseEnergyLoss_t_42760694399156
// MI455X (gfx1250) — matrix operand order repaired and checked by reading, not compiled, not run on hardware
//
#include <hip/hip_runtime.h>

// MI455X / gfx1250 implementation of dense_energy_loss.
// Strategy: the loss only needs Gram dot-products <seg_roi(p), seg_roi(p+off)>
// over K=21 channels -> v_wmma_f32_16x16x32_f16 (channels padded to 32).
// Bilateral weights collapse to ONE exp per tap:
//   w = [dx^2+dy^2 <= 49] * exp( -(dx^2+dy^2)/5000 - (gsc^2/450)*L1(raw guide)^2 )
// (spatial gaussian folded into the exponent; guide normalization gsc folded
//  into the color coefficient since |gsc*a - gsc*b| = gsc*|a-b|).

typedef __attribute__((ext_vector_type(16))) _Float16 v16h;
typedef __attribute__((ext_vector_type(8)))  float    v8f;

#define NIMG 16
#define HS   160
#define WS   160
#define KCH  21
#define KP   32   // channel count in the global f16 buffer
#define KPAD 40   // channel slot in LDS (80B: 16B-aligned, bank-conflict free)
#define RAD  7

union Frag { v16h h; uint4 q[2]; unsigned u[8]; };

__device__ __forceinline__ int refl(int v, int n) {
  // jnp.pad 'reflect' (mirror without repeating the edge)
  v = (v < 0) ? -v : v;
  v = (v >= n) ? (2 * n - 2 - v) : v;
  return v;
}

// --- CDNA5 async global->LDS copy (ASYNCcnt-tracked), with safe fallback ----
#if __has_builtin(__builtin_amdgcn_global_load_async_to_lds_b128) && \
    __has_builtin(__builtin_amdgcn_global_load_async_to_lds_b32)
#define USE_ASYNC_LDS 1
typedef int i4 __attribute__((vector_size(16)));  // matches builtin pointee
typedef __attribute__((address_space(1))) i4*  g128_t;
typedef __attribute__((address_space(3))) i4*  l128_t;
typedef __attribute__((address_space(1))) int* g32_t;
typedef __attribute__((address_space(3))) int* l32_t;
__device__ __forceinline__ g128_t to_g128(const void* p) {
  return (g128_t)(uintptr_t)p;            // global AS1: full 64-bit address
}
__device__ __forceinline__ l128_t to_l128(void* p) {
  return (l128_t)(unsigned)(uintptr_t)p;  // LDS AS3: low 32 bits = LDS addr
}
__device__ __forceinline__ g32_t to_g32(const void* p) {
  return (g32_t)(uintptr_t)p;
}
__device__ __forceinline__ l32_t to_l32(void* p) {
  return (l32_t)(unsigned)(uintptr_t)p;
}
#endif

// ---------------------------------------------------------------------------
// Kernel 0: zero the per-image guide-max accumulators.
// ---------------------------------------------------------------------------
__global__ void k_init(unsigned* __restrict__ gmaxG) {
  if (threadIdx.x < NIMG) gmaxG[threadIdx.x] = 0u;
}

// ---------------------------------------------------------------------------
// Kernel 1: downsample + derive per-pixel data.
//   - guideG : raw img_s values (3 ch, f32); normalization decided later
//   - segG   : seg_roi = 2x2-mean(seg) * roi, f16, padded to 32 channels
//   - gateG  : max(roi - seg_max, 0) or 1 where label==255
//   - gmaxG  : per-image max of raw guide (for the gmax>1 -> /255 rule)
// ---------------------------------------------------------------------------
__global__ void __launch_bounds__(256) k_prep(
    const float* __restrict__ images, const float* __restrict__ segm,
    const float* __restrict__ rois, const int* __restrict__ labels,
    _Float16* __restrict__ segG, float* __restrict__ guideG,
    float* __restrict__ gateG, unsigned* __restrict__ gmaxG) {
  __shared__ float redmax[256];
  const int n   = blockIdx.x / 100;
  const int pix = (blockIdx.x % 100) * 256 + threadIdx.x;  // 0..25599
  const int i = pix / WS, j = pix % WS;
  const int i2 = 2 * i, j2 = 2 * j;
  const size_t p = ((size_t)n * HS + i) * WS + j;

  float lmax = 0.f;
#pragma unroll
  for (int c = 0; c < 3; ++c) {
    float g = images[(((size_t)n * 3 + c) * 320 + i2) * 320 + j2];
    guideG[p * 3 + c] = g;
    lmax = fmaxf(lmax, g);
  }

  const float roi = rois[((size_t)n * 320 + i2) * 320 + j2];

  union { _Float16 h[KP]; uint4 q[4]; } outv;
  const float2* seg2 = (const float2*)segm;  // 2 consecutive cols per load
  float smax = -1e30f;
#pragma unroll
  for (int k = 0; k < KCH; ++k) {
    size_t b0 = (((size_t)n * KCH + k) * 320 + i2) * 160 + j;  // float2 index
    float2 a  = seg2[b0];
    float2 bb = seg2[b0 + 160];
    float s = 0.25f * (a.x + a.y + bb.x + bb.y);  // 2x2 mean pool
    smax = fmaxf(smax, s);
    outv.h[k] = (_Float16)(s * roi);
  }
#pragma unroll
  for (int k = KCH; k < KP; ++k) outv.h[k] = (_Float16)0.f;
  uint4* dst = (uint4*)(segG + p * KP);
  dst[0] = outv.q[0]; dst[1] = outv.q[1]; dst[2] = outv.q[2]; dst[3] = outv.q[3];

  const int lab = labels[((size_t)n * 320 + i2) * 320 + j2];
  gateG[p] = (lab == 255) ? 1.f : fmaxf(roi - smax, 0.f);

  // block-level max, then one order-independent atomicMax per block
  redmax[threadIdx.x] = lmax;
  __syncthreads();
  for (int s = 128; s > 0; s >>= 1) {
    if (threadIdx.x < s)
      redmax[threadIdx.x] = fmaxf(redmax[threadIdx.x], redmax[threadIdx.x + s]);
    __syncthreads();
  }
  if (threadIdx.x == 0) atomicMax(gmaxG + n, __float_as_uint(redmax[0]));
}

// ---------------------------------------------------------------------------
// Kernel 2: bilateral Gram accumulation. One block = 16(x) x 8(y) output tile,
// 8 wave32s, one 16-pixel row per wave; 30 WMMAs/wave cover all 15x15 offsets.
// ---------------------------------------------------------------------------
__global__ void __launch_bounds__(256) k_main(
    const _Float16* __restrict__ segG, const float* __restrict__ guideG,
    const float* __restrict__ gateG, const unsigned* __restrict__ gmaxG,
    float* __restrict__ partials) {
  __shared__ _Float16 segT[22][32][KPAD];  // halo rows y0-7..y0+14, cols x0-8..x0+23
  __shared__ float    guiT[22][32][3];     // RAW guide halo
  __shared__ float    red[16];

  const int b  = blockIdx.x;
  const int n  = b / 200;
  const int t  = b % 200;
  const int y0 = (t / 10) * 8;
  const int x0 = (t % 10) * 16;
  const int tid = threadIdx.x;

  const float gm  = __uint_as_float(gmaxG[n]);
  const float gsc = (gm > 1.f) ? (1.f / 255.f) : 1.f;
  const float c1  = -1.f / 5000.f;          // -1/(2*(SIGMA_XY*SCALE)^2)
  const float c2  = -(gsc * gsc) / 450.f;   // -gsc^2/(2*SIGMA_RGB^2)

  for (int site = tid; site < 22 * 32; site += 256) {
    int lr = site >> 5, lc = site & 31;
    int gy = refl(y0 - RAD + lr, HS);
    int gx = refl(x0 - 8 + lc, WS);
    size_t p = ((size_t)n * HS + gy) * WS + gx;
#ifdef USE_ASYNC_LDS
    const _Float16* sp = segG + p * KP;
    _Float16* dp = &segT[lr][lc][0];
    // IOFFSET is added to BOTH the global and LDS addresses (ISA 15.18.3)
    __builtin_amdgcn_global_load_async_to_lds_b128(to_g128(sp), to_l128(dp), 0, 0);
    __builtin_amdgcn_global_load_async_to_lds_b128(to_g128(sp), to_l128(dp), 16, 0);
    __builtin_amdgcn_global_load_async_to_lds_b128(to_g128(sp), to_l128(dp), 32, 0);
    __builtin_amdgcn_global_load_async_to_lds_b128(to_g128(sp), to_l128(dp), 48, 0);
    const float* gp = guideG + p * 3;
    float* gq = &guiT[lr][lc][0];
    __builtin_amdgcn_global_load_async_to_lds_b32(to_g32(gp), to_l32(gq), 0, 0);
    __builtin_amdgcn_global_load_async_to_lds_b32(to_g32(gp), to_l32(gq), 4, 0);
    __builtin_amdgcn_global_load_async_to_lds_b32(to_g32(gp), to_l32(gq), 8, 0);
#else
    const uint4* sp = (const uint4*)(segG + p * KP);
    uint4* dp = (uint4*)&segT[lr][lc][0];
    dp[0] = sp[0]; dp[1] = sp[1]; dp[2] = sp[2]; dp[3] = sp[3];
    guiT[lr][lc][0] = guideG[p * 3 + 0];
    guiT[lr][lc][1] = guideG[p * 3 + 1];
    guiT[lr][lc][2] = guideG[p * 3 + 2];
#endif
  }
#ifdef USE_ASYNC_LDS
#if __has_builtin(__builtin_amdgcn_s_wait_asynccnt)
  __builtin_amdgcn_s_wait_asynccnt(0);
#else
  asm volatile("s_wait_asynccnt 0x0" ::: "memory");
#endif
#endif
  __syncthreads();

  const int wid = tid >> 5;
  const int lane = tid & 31;
  const int hi = lane >> 4;
  const int l16 = lane & 15;
  const int ly = wid + RAD;  // wave's output row in the halo
  const int y  = y0 + wid;

  // A fragment: 16 pixels (M) x 32 channels (K), 16-bit A layout:
  // lane l16 -> row M=l16; lanes 0-15 hold K 0..7 & 16..23, lanes 16-31: 8..15 & 24..31.
  Frag A;
  {
    const uint4* ap = (const uint4*)&segT[ly][l16 + 8][0];
    A.q[0] = ap[hi];       // ch 8*hi    .. 8*hi+7
    A.q[1] = ap[2 + hi];   // ch 8*hi+16 .. 8*hi+23
  }
  // raw guide of this lane's 8 D-matrix rows (D layout: M = r + 8*hi)
  float gA[8][3];
#pragma unroll
  for (int r = 0; r < 8; ++r) {
    int lcM = r + 8 * hi + 8;
    gA[r][0] = guiT[ly][lcM][0];
    gA[r][1] = guiT[ly][lcM][1];
    gA[r][2] = guiT[ly][lcM][2];
  }

  float accN[8] = {0, 0, 0, 0, 0, 0, 0, 0};
  float accD[8] = {0, 0, 0, 0, 0, 0, 0, 0};

#pragma unroll 1
  for (int dy = -RAD; dy <= RAD; ++dy) {
    const int lyb = ly + dy;
    const int dy2 = dy * dy;
#pragma unroll
    for (int side = 0; side < 2; ++side) {
      const int lc = l16 + 16 * side;  // neighbor column (N = l16)
      // B fragment: 32(K) x 16(N); lanes 0-15 K=0..7 & 16..23, lanes 16-31 K=8..15 & 24..31.
      Frag B;
      const uint4* bp = (const uint4*)&segT[lyb][lc][0];
      B.q[0] = bp[hi];
      B.q[1] = bp[2 + hi];
      v8f cz = {};
      v8f d = __builtin_amdgcn_wmma_f32_16x16x32_f16(
          false, A.h, false, B.h, (short)0, cz, false, false);
      const float gb0 = guiT[lyb][lc][0];
      const float gb1 = guiT[lyb][lc][1];
      const float gb2 = guiT[lyb][lc][2];
      // dx for D row r: neighbor x = x0-8+l16+16*side, pixel x = x0+r+8*hi
      const int dxbase = l16 + 16 * side - 8 - 8 * hi;
#pragma unroll
      for (int r = 0; r < 8; ++r) {
        const int dx = dxbase - r;
        const int r2 = dx * dx + dy2;
        float dd = fabsf(gA[r][0] - gb0) + fabsf(gA[r][1] - gb1) +
                   fabsf(gA[r][2] - gb2);
        float e = fmaf(dd * dd, c2, (float)r2 * c1);
        float w = __expf(e);               // branchless: single exp per tap
        w = (r2 <= RAD * RAD) ? w : 0.f;   // disk mask -> v_cndmask
        accN[r] = fmaf(w, d[r], accN[r]);
        accD[r] += w;
      }
    }
  }

  // reduce over the 16 N-lanes of each half-wave (fixed order -> deterministic)
#pragma unroll
  for (int r = 0; r < 8; ++r) {
#pragma unroll
    for (int o = 1; o < 16; o <<= 1) {
      accN[r] += __shfl_xor(accN[r], o, 16);
      accD[r] += __shfl_xor(accD[r], o, 16);
    }
  }
  if (l16 == 0) {
    float part = 0.f;
#pragma unroll
    for (int r = 0; r < 8; ++r) {
      int xm = x0 + r + 8 * hi;
      float gate = gateG[((size_t)n * HS + y) * WS + xm];
      part += gate * accN[r] / accD[r];  // accD >= 1 (center tap weight == 1)
    }
    red[wid * 2 + hi] = part;
  }
  __syncthreads();
  if (tid == 0) {
    float s = 0.f;
#pragma unroll
    for (int k = 0; k < 16; ++k) s += red[k];
    partials[b] = s;
  }
}

// ---------------------------------------------------------------------------
// Kernel 3: deterministic final reduction + scaling.
// ---------------------------------------------------------------------------
__global__ void __launch_bounds__(256) k_final(const float* __restrict__ partials,
                                               int np, float* __restrict__ out) {
  __shared__ float red[256];
  float s = 0.f;
  for (int i = threadIdx.x; i < np; i += 256) s += partials[i];
  red[threadIdx.x] = s;
  __syncthreads();
  for (int o = 128; o > 0; o >>= 1) {
    if (threadIdx.x < o) red[threadIdx.x] += red[threadIdx.x + o];
    __syncthreads();
  }
  if (threadIdx.x == 0) out[0] = red[0] * (-1e-7f / (float)NIMG);
}

// ---------------------------------------------------------------------------
extern "C" void kernel_launch(void* const* d_in, const int* in_sizes, int n_in,
                              void* d_out, int out_size, void* d_ws, size_t ws_size,
                              hipStream_t stream) {
  (void)in_sizes; (void)n_in; (void)out_size; (void)ws_size;
  const float* images = (const float*)d_in[0];
  const float* segm   = (const float*)d_in[1];
  const float* rois   = (const float*)d_in[2];
  const int*   labels = (const int*)d_in[3];

  char* ws = (char*)d_ws;
  size_t off = 0;
  auto alloc = [&](size_t bytes) {
    off = (off + 255) & ~(size_t)255;
    size_t o = off;
    off += bytes;
    return o;
  };
  _Float16* segG   = (_Float16*)(ws + alloc((size_t)NIMG * HS * WS * KP * sizeof(_Float16)));  // 26.2 MB
  float*    guideG = (float*)(ws + alloc((size_t)NIMG * HS * WS * 3 * sizeof(float)));         // 4.9 MB
  float*    gateG  = (float*)(ws + alloc((size_t)NIMG * HS * WS * sizeof(float)));             // 1.6 MB
  unsigned* gmaxG  = (unsigned*)(ws + alloc(NIMG * sizeof(unsigned)));
  float*    parts  = (float*)(ws + alloc(3200 * sizeof(float)));

  k_init<<<1, 64, 0, stream>>>(gmaxG);
  k_prep<<<NIMG * 100, 256, 0, stream>>>(images, segm, rois, labels,
                                         segG, guideG, gateG, gmaxG);
  k_main<<<3200, 256, 0, stream>>>(segG, guideG, gateG, gmaxG, parts);
  k_final<<<1, 256, 0, stream>>>(parts, 3200, (float*)d_out);
}
